// FeedForward_79190607004010
// MI455X (gfx1250) — matrix operand order repaired and checked by reading, not compiled, not run on hardware
//
#include <hip/hip_runtime.h>
#include <hip/hip_bf16.h>
#include <math.h>

typedef __bf16 bf16;
typedef __attribute__((ext_vector_type(16))) __bf16 v16bf;
typedef __attribute__((ext_vector_type(8)))  float  v8f;

#define DIM    1024
#define N1     4096
#define HID    2048
#define SEQ    2048
#define NBATCH 8
#define NROWS  (NBATCH * SEQ)          // 16384
#define NSHIFT 7
#define CSIZE  (HID / (NSHIFT + 1))    // 256
#define LN_EPS    1e-5f
#define SHIFT_EPS 1e-5f

// LDS row pitch for staged tiles: 32 data elements + 8 pad = 40 bf16 = 80B.
// 80B = 20 banks; 20*r mod 64 covers all multiples of 4 -> every 16B fragment
// read hits each of the 64 banks exactly twice per 32-lane access (optimal).
#define LPITCH 40

// ---------------------------------------------------------------------------
// CDNA5 async global->LDS copy (ASYNCcnt path), 16B per lane.
// ---------------------------------------------------------------------------
__device__ __forceinline__ void async_copy_b128(unsigned lds_byte_off, const void* gaddr) {
    asm volatile("global_load_async_to_lds_b128 %0, %1, off"
                 :: "v"(lds_byte_off), "v"(gaddr)
                 : "memory");
}
__device__ __forceinline__ void wait_async0() {
    asm volatile("s_wait_asynccnt 0x0" ::: "memory");
}
__device__ __forceinline__ unsigned lds_off32(const void* p) {
    // generic pointer to __shared__: low 32 bits == LDS byte offset
    return (unsigned)(uintptr_t)p;
}

// ---------------------------------------------------------------------------
// Weight convert + transpose: W[K][N] f32 (row-major) -> WT[N][K] bf16
// ---------------------------------------------------------------------------
__global__ __launch_bounds__(256) void convT_kernel(const float* __restrict__ W,
                                                    bf16* __restrict__ WT,
                                                    int K, int N) {
    size_t idx = (size_t)blockIdx.x * blockDim.x + threadIdx.x;
    if (idx >= (size_t)K * N) return;
    int n = (int)(idx % N);
    int k = (int)(idx / N);
    WT[(size_t)n * K + k] = (bf16)W[idx];
}

// ---------------------------------------------------------------------------
// Row LayerNorm. WIDTH elements per row, 256 threads/row.
// ---------------------------------------------------------------------------
template <int WIDTH, bool TOBF>
__global__ __launch_bounds__(256) void ln_kernel(const float* __restrict__ X,
                                                 const float* __restrict__ w,
                                                 const float* __restrict__ b,
                                                 void* __restrict__ out) {
    constexpr int PER = WIDTH / 256;
    int row = blockIdx.x;
    const float* xr = X + (size_t)row * WIDTH;

    float v[PER];
    float s = 0.f, s2 = 0.f;
#pragma unroll
    for (int i = 0; i < PER; ++i) {
        v[i] = xr[threadIdx.x + i * 256];
        s += v[i];
        s2 += v[i] * v[i];
    }
    __shared__ float r1[256];
    __shared__ float r2[256];
    r1[threadIdx.x] = s;
    r2[threadIdx.x] = s2;
    __syncthreads();
    for (int off = 128; off > 0; off >>= 1) {
        if (threadIdx.x < (unsigned)off) {
            r1[threadIdx.x] += r1[threadIdx.x + off];
            r2[threadIdx.x] += r2[threadIdx.x + off];
        }
        __syncthreads();
    }
    float mu  = r1[0] * (1.0f / WIDTH);
    float var = r2[0] * (1.0f / WIDTH) - mu * mu;
    float rs  = rsqrtf(var + LN_EPS);

#pragma unroll
    for (int i = 0; i < PER; ++i) {
        int c = threadIdx.x + i * 256;
        float y = (v[i] - mu) * rs * w[c] + b[c];
        if (TOBF)
            ((bf16*)out)[(size_t)row * WIDTH + c] = (bf16)y;
        else
            ((float*)out)[(size_t)row * WIDTH + c] = y;
    }
}

// ---------------------------------------------------------------------------
// In-place cumsum over sequence dim for channels [0, NSHIFT*CSIZE) of gate.
// ---------------------------------------------------------------------------
__global__ __launch_bounds__(256) void cumsum_kernel(float* __restrict__ g) {
    int idx = blockIdx.x * blockDim.x + threadIdx.x;  // < NBATCH * NSHIFT*CSIZE
    int bi = idx / (NSHIFT * CSIZE);
    int c  = idx % (NSHIFT * CSIZE);
    size_t base = (size_t)bi * SEQ * HID + c;
    float s = 0.f;
    for (int t = 0; t < SEQ; ++t) {
        size_t o = base + (size_t)t * HID;
        s += g[o];
        g[o] = s;
    }
}

// ---------------------------------------------------------------------------
// Token shift from cumsum differences; last chunk passes raw gate through.
// ---------------------------------------------------------------------------
__global__ __launch_bounds__(256) void shift_kernel(const float* __restrict__ g,
                                                    bf16* __restrict__ out) {
    size_t idx = (size_t)blockIdx.x * blockDim.x + threadIdx.x;
    if (idx >= (size_t)NROWS * HID) return;
    int c = (int)(idx % HID);
    size_t row = idx / HID;          // b*SEQ + t
    int t = (int)(row % SEQ);
    float r;
    if (c >= NSHIFT * CSIZE) {
        r = g[idx];
    } else {
        int i = c / CSIZE;
        int a = 1 << i;
        float num = 0.f, sd = 0.f;
        if (t >= a)     { num += g[(row - (size_t)a) * HID + c];     sd += (float)(t - a); }
        if (t >= 2 * a) { num -= g[(row - (size_t)2*a) * HID + c];   sd -= (float)(t - 2*a); }
        r = num / (sd + SHIFT_EPS);
    }
    out[idx] = (bf16)r;
}

// ---------------------------------------------------------------------------
// LDS fragment loads (WMMA operand layouts, see round-0 analysis).
// ---------------------------------------------------------------------------
__device__ __forceinline__ v16bf lds_a_frag(const bf16* Ab, int rowbase, int lane) {
    int row = rowbase + (lane & 15);
    int ke  = (lane & 16) ? 8 : 0;          // K element offset within 32
    const bf16* p = Ab + row * LPITCH + ke;
    union { uint4 u[2]; v16bf v; } u;
    u.u[0] = *(const uint4*)(const void*)p;        // K +0..7
    u.u[1] = *(const uint4*)(const void*)(p + 16); // K +16..23
    return u.v;
}
__device__ __forceinline__ v16bf lds_b_frag(const bf16* Bb, int rowbase, int lane) {
    int row = rowbase + (lane & 15);        // N column (BT row)
    int ke  = (lane & 16) ? 8 : 0;
    const bf16* p = Bb + row * LPITCH + ke;
    union { uint4 u[2]; v16bf v; } u;
    u.u[0] = *(const uint4*)(const void*)p;        // K +0..7
    u.u[1] = *(const uint4*)(const void*)(p + 16); // K +16..23
    return u.v;
}

// ---------------------------------------------------------------------------
// LDS-staged bf16 WMMA GEMM, f32 accumulate, async double-buffered operands.
// Block = 256 threads = 8 waves (2x4). Block tile 128x256, wave tile 64x64
// (16 WMMA accumulators). K step = 32.
// EPI 0: bias + exact GELU, split columns -> h bf16 / gate f32
// EPI 1: bias, multiply by h (bf16), write bf16
// EPI 2: bias, write f32
// ---------------------------------------------------------------------------
template <int EPI>
__global__ __launch_bounds__(256, 1) void gemm_wmma_lds_kernel(
    const bf16*  __restrict__ A,
    const bf16*  __restrict__ BT,
    const float* __restrict__ bias,
    int M, int N, int K,
    const bf16*  __restrict__ hmul,
    bf16*        __restrict__ outB,
    float*       __restrict__ outF) {

    __shared__ bf16 Asm[2][128 * LPITCH];   // 2 x 10 KB
    __shared__ bf16 Bsm[2][256 * LPITCH];   // 2 x 20 KB

    const int tid  = threadIdx.x;
    const int lane = tid & 31;
    const int wid  = tid >> 5;
    const int wm   = wid >> 2;              // 0..1
    const int wn   = wid & 3;               // 0..3

    const int m0 = blockIdx.y * 128;        // block row base
    const int n0 = blockIdx.x * 256;        // block col base

    // ---- async copy issue for one K-step into buffer `buf` -----------------
    auto issue = [&](int k0, int buf) {
        // A tile: 128 rows x 64B  -> 512 x 16B pieces, 2 per thread
#pragma unroll
        for (int i = 0; i < 2; ++i) {
            int p   = tid + 256 * i;
            int row = p >> 2, seg = p & 3;
            const bf16* g = A + (size_t)(m0 + row) * K + k0 + seg * 8;
            async_copy_b128(lds_off32(&Asm[buf][row * LPITCH + seg * 8]), g);
        }
        // B tile: 256 cols x 64B -> 1024 x 16B pieces, 4 per thread
#pragma unroll
        for (int i = 0; i < 4; ++i) {
            int p   = tid + 256 * i;
            int row = p >> 2, seg = p & 3;
            const bf16* g = BT + (size_t)(n0 + row) * K + k0 + seg * 8;
            async_copy_b128(lds_off32(&Bsm[buf][row * LPITCH + seg * 8]), g);
        }
    };

    v8f acc[4][4];
#pragma unroll
    for (int i = 0; i < 4; ++i)
#pragma unroll
        for (int j = 0; j < 4; ++j)
#pragma unroll
            for (int r = 0; r < 8; ++r) acc[i][j][r] = 0.f;

    const int nsteps = K >> 5;
    issue(0, 0);

    for (int s = 0; s < nsteps; ++s) {
        wait_async0();          // my copies into buf (s&1) have landed in LDS
        __syncthreads();        // everyone's copies landed; prev reads done
        const int cur = s & 1;
        if (s + 1 < nsteps) issue((s + 1) << 5, cur ^ 1);

        const bf16* Ab = Asm[cur];
        const bf16* Bb = Bsm[cur];

        v16bf af[4];
#pragma unroll
        for (int mt = 0; mt < 4; ++mt)
            af[mt] = lds_a_frag(Ab, wm * 64 + mt * 16, lane);
#pragma unroll
        for (int nt = 0; nt < 4; ++nt) {
            v16bf bfr = lds_b_frag(Bb, wn * 64 + nt * 16, lane);
#pragma unroll
            for (int mt = 0; mt < 4; ++mt)
                acc[mt][nt] = __builtin_amdgcn_wmma_f32_16x16x32_bf16(
                    false, af[mt], false, bfr, (short)0, acc[mt][nt], false, false);
        }
    }

    // C/D layout: element r -> M = tile_m + r + (lane>=16 ? 8:0), N = tile_n + lane%16
    const int ncol = lane & 15;
    const int radd = (lane & 16) ? 8 : 0;
#pragma unroll
    for (int mt = 0; mt < 4; ++mt) {
#pragma unroll
        for (int nt = 0; nt < 4; ++nt) {
            int n = n0 + wn * 64 + nt * 16 + ncol;
            float bv = bias[n];
#pragma unroll
            for (int r = 0; r < 8; ++r) {
                int m = m0 + wm * 64 + mt * 16 + radd + r;
                float v = acc[mt][nt][r] + bv;
                if (EPI == 0) {
                    float gg = 0.5f * v * (1.0f + erff(v * 0.7071067811865476f));
                    if (n < HID) outB[(size_t)m * HID + n] = (bf16)gg;
                    else         outF[(size_t)m * HID + (n - HID)] = gg;
                } else if (EPI == 1) {
                    float hv = (float)hmul[(size_t)m * N + n];
                    outB[(size_t)m * N + n] = (bf16)(v * hv);
                } else {
                    outF[(size_t)m * N + n] = v;
                }
            }
        }
    }
}

// ---------------------------------------------------------------------------
// Host launcher
// ---------------------------------------------------------------------------
extern "C" void kernel_launch(void* const* d_in, const int* in_sizes, int n_in,
                              void* d_out, int out_size, void* d_ws, size_t ws_size,
                              hipStream_t stream) {
    const float* x      = (const float*)d_in[0];
    const float* ln_w   = (const float*)d_in[1];
    const float* ln_b   = (const float*)d_in[2];
    const float* w_in   = (const float*)d_in[3];
    const float* b_in   = (const float*)d_in[4];
    const float* gln_w  = (const float*)d_in[5];
    const float* gln_b  = (const float*)d_in[6];
    const float* w_gate = (const float*)d_in[7];
    const float* b_gate = (const float*)d_in[8];
    const float* w_out  = (const float*)d_in[9];
    const float* b_out  = (const float*)d_in[10];
    float* out = (float*)d_out;

    char* ws = (char*)d_ws;
    size_t off = 0;
    auto alloc = [&](size_t bytes) -> void* {
        void* p = ws + off;
        off = (off + bytes + 255) & ~(size_t)255;
        return p;
    };
    bf16*  lnx    = (bf16*) alloc((size_t)NROWS * DIM * sizeof(bf16));
    bf16*  w_inT  = (bf16*) alloc((size_t)DIM * N1 * sizeof(bf16));
    bf16*  w_gT   = (bf16*) alloc((size_t)HID * HID * sizeof(bf16));
    bf16*  w_oT   = (bf16*) alloc((size_t)HID * DIM * sizeof(bf16));
    bf16*  hbuf   = (bf16*) alloc((size_t)NROWS * HID * sizeof(bf16));
    float* gate   = (float*)alloc((size_t)NROWS * HID * sizeof(float));
    bf16*  shiftb = (bf16*) alloc((size_t)NROWS * HID * sizeof(bf16));
    bf16*  h2     = (bf16*) alloc((size_t)NROWS * HID * sizeof(bf16));

    {
        size_t n = (size_t)DIM * N1;
        convT_kernel<<<(unsigned)((n + 255) / 256), 256, 0, stream>>>(w_in, w_inT, DIM, N1);
    }
    {
        size_t n = (size_t)HID * HID;
        convT_kernel<<<(unsigned)((n + 255) / 256), 256, 0, stream>>>(w_gate, w_gT, HID, HID);
    }
    {
        size_t n = (size_t)HID * DIM;
        convT_kernel<<<(unsigned)((n + 255) / 256), 256, 0, stream>>>(w_out, w_oT, HID, DIM);
    }

    // 1) LN(x) -> bf16
    ln_kernel<DIM, true><<<NROWS, 256, 0, stream>>>(x, ln_w, ln_b, lnx);

    // 2) GEMM1 + bias + GELU, split h / gate
    {
        dim3 grid(N1 / 256, NROWS / 128);
        gemm_wmma_lds_kernel<0><<<grid, 256, 0, stream>>>(
            lnx, w_inT, b_in, NROWS, N1, DIM, nullptr, hbuf, gate);
    }

    // 3) LN(gate) in place (f32)
    ln_kernel<HID, false><<<NROWS, 256, 0, stream>>>(gate, gln_w, gln_b, gate);

    // 4) cumsum over sequence for first 7*256 channels (in place)
    cumsum_kernel<<<(NBATCH * NSHIFT * CSIZE) / 256, 256, 0, stream>>>(gate);

    // 5) shifted differences -> bf16
    {
        size_t n = (size_t)NROWS * HID;
        shift_kernel<<<(unsigned)((n + 255) / 256), 256, 0, stream>>>(gate, shiftb);
    }

    // 6) GEMM2 + bias, multiply by h -> bf16
    {
        dim3 grid(HID / 256, NROWS / 128);
        gemm_wmma_lds_kernel<1><<<grid, 256, 0, stream>>>(
            shiftb, w_gT, b_gate, NROWS, HID, HID, hbuf, h2, nullptr);
    }

    // 7) GEMM3 + bias -> f32 output
    {
        dim3 grid(DIM / 256, NROWS / 128);
        gemm_wmma_lds_kernel<2><<<grid, 256, 0, stream>>>(
            h2, w_oT, b_out, NROWS, DIM, HID, nullptr, nullptr, out);
    }
}
